// MultiHeadSelfAttention_21208548508374
// MI455X (gfx1250) — matrix operand order repaired and checked by reading, not compiled, not run on hardware
//
#include <hip/hip_runtime.h>
#include <hip/hip_bf16.h>

#define S_LEN 2048
#define BATCH 2
#define DM    1024
#define NH    16
#define DH    64

typedef __attribute__((ext_vector_type(16))) __bf16 v16bf;
typedef __attribute__((ext_vector_type(8)))  float  v8f;

union FragU { unsigned int u[8]; v16bf v; };

__device__ __forceinline__ v8f v8f_zero() {
    v8f z;
#pragma unroll
    for (int i = 0; i < 8; ++i) z[i] = 0.0f;
    return z;
}

// A-fragment (16x32 MxK, bf16): lane m = lane&15, element-pair j -> K pair
// kpair(j) = j + 8*(lane>>4); pairs contiguous in memory.
__device__ __forceinline__ v16bf load_a_frag(const __hip_bfloat16* rowptr, int half) {
    FragU f;
#pragma unroll
    for (int j = 0; j < 8; ++j) {
        int kp = j + 8 * half;
        f.u[j] = *(const unsigned int*)(rowptr + 2 * kp);
    }
    return f.v;
}

// B-fragment (32x16 KxN, bf16): lane n = lane&15, k = 16*(lane>>4) + 2j(+1).
// rowptr points at this lane's column-n K-run (works for global or LDS).
__device__ __forceinline__ v16bf load_b_frag(const __hip_bfloat16* rowptr, int half) {
    FragU f;
#pragma unroll
    for (int j = 0; j < 8; ++j) {
        f.u[j] = *(const unsigned int*)(rowptr + 16 * half + 2 * j);
    }
    return f.v;
}

__device__ __forceinline__ v8f wmma_bf16(v16bf a, v16bf b, v8f c) {
    return __builtin_amdgcn_wmma_f32_16x16x32_bf16(false, a, false, b, (short)0, c, false, false);
}

// ------------------------------------------------------------------ TDM
// Stage a 64(N-rows) x 32(K) bf16 tile of a [1024x1024] row-major matrix
// into LDS via the Tensor Data Mover (async DMA, TENSORcnt-tracked).
#if defined(__gfx1250__) && __has_builtin(__builtin_amdgcn_tensor_load_to_lds)
#define HAVE_TDM 1
#else
#define HAVE_TDM 0
#endif

typedef __attribute__((ext_vector_type(4))) unsigned tdm_v4u;
typedef __attribute__((ext_vector_type(8))) int      tdm_v8i;
typedef __attribute__((ext_vector_type(4))) int      tdm_v4i;

__device__ __forceinline__ void tdm_load_tile64x32(const __hip_bfloat16* gsrc,
                                                   __hip_bfloat16* lds_dst) {
#if HAVE_TDM
    unsigned long long ga = (unsigned long long)(size_t)gsrc;
    unsigned ldsoff = (unsigned)(size_t)(void*)lds_dst;  // low 32 bits = LDS byte addr
    tdm_v4u g0;
    g0[0] = 1u;                                    // count=1, user descriptor
    g0[1] = ldsoff;                                // lds_addr
    g0[2] = (unsigned)(ga & 0xffffffffu);          // global_addr[31:0]
    g0[3] = (unsigned)((ga >> 32) & 0x01ffffffu)   // global_addr[56:32]
            | 0x80000000u;                         // type=2 ("image")
    tdm_v8i g1;
    g1[0] = 0x00010000;    // wg_mask=0, data_size=1 (2 bytes)
    g1[1] = 0x04000000;    // tensor_dim0 = 1024 (bits[79:48], low16 here)
    g1[2] = 0x04000000;    // tensor_dim0 hi=0 | tensor_dim1=1024 low16
    g1[3] = (32 << 16);    // tensor_dim1 hi=0 | tile_dim0 = 32 (K)
    g1[4] = 64;            // tile_dim1 = 64 (rows), tile_dim2 = 0
    g1[5] = 1024;          // tensor_dim0_stride[31:0] = 1024 elements
    g1[6] = 0;             // stride hi / dim1_stride lo
    g1[7] = 0;
    tdm_v4i g2 = {0, 0, 0, 0};
    tdm_v4i g3 = {0, 0, 0, 0};
    tdm_v8i g4 = {0, 0, 0, 0, 0, 0, 0, 0};
    __builtin_amdgcn_tensor_load_to_lds(g0, g1, g2, g3, g4, 0);
#else
    // Fallback: synchronous copy by the issuing wave (32 lanes).
    int lane = threadIdx.x & 31;
    for (int r = lane; r < 64; r += 32) {
        const uint4* s = (const uint4*)(gsrc + (size_t)r * DM);
        uint4* d = (uint4*)(lds_dst + r * 32);
#pragma unroll
        for (int c = 0; c < 4; ++c) d[c] = s[c];
    }
#endif
}

__device__ __forceinline__ void tdm_wait0() {
#if HAVE_TDM
#if __has_builtin(__builtin_amdgcn_s_wait_tensorcnt)
    __builtin_amdgcn_s_wait_tensorcnt(0);
#else
    asm volatile("s_wait_tensorcnt 0x0" ::: "memory");
#endif
#endif
}
__device__ __forceinline__ void tdm_wait1() {
#if HAVE_TDM
#if __has_builtin(__builtin_amdgcn_s_wait_tensorcnt)
    __builtin_amdgcn_s_wait_tensorcnt(1);
#else
    asm volatile("s_wait_tensorcnt 0x1" ::: "memory");
#endif
#endif
}

// ---------------------------------------------------------------- converts
__global__ void cvt_f32_bf16(const float* __restrict__ src,
                             __hip_bfloat16* __restrict__ dst, int n) {
    int i = blockIdx.x * blockDim.x + threadIdx.x;
    int stride = gridDim.x * blockDim.x;
    for (; i < n; i += stride) dst[i] = __float2bfloat16(src[i]);
}

// ---------------------------------------------------------------- QKV GEMM + RoPE
// grid: (M/128, N/64, 3); block 256 (8 waves). Wave w -> 16x64 output tile.
// B (weight) tile double-buffered in LDS via TDM; A frags streamed from global.
// Q,K stored [b,h,s,d] with RoPE applied; V stored transposed [b,h,d,s].
__global__ __launch_bounds__(256) void qkv_rope_kernel(
    const __hip_bfloat16* __restrict__ x,
    const __hip_bfloat16* __restrict__ wq,
    const __hip_bfloat16* __restrict__ wk,
    const __hip_bfloat16* __restrict__ wv,
    const int* __restrict__ posn,
    __hip_bfloat16* __restrict__ qout,
    __hip_bfloat16* __restrict__ kout,
    __hip_bfloat16* __restrict__ vtout) {
    __shared__ __align__(16) __hip_bfloat16 bt[2][64][32];
    const int lane = threadIdx.x & 31;
    const int half = lane >> 4;
    const int l16  = lane & 15;
    const int w    = threadIdx.x >> 5;
    const int mt   = blockIdx.x * 128 + w * 16;
    const int nt   = blockIdx.y * 64;
    const int z    = blockIdx.z;
    const __hip_bfloat16* W = (z == 0) ? wq : ((z == 1) ? wk : wv);
    const __hip_bfloat16* wbase = W + (size_t)nt * DM;

    v8f acc[4];
#pragma unroll
    for (int g = 0; g < 4; ++g) acc[g] = v8f_zero();

    const __hip_bfloat16* arow = x + (size_t)(mt + l16) * DM;

    const int NSTEPS = DM / 32;
    if (w == 0) tdm_load_tile64x32(wbase, &bt[0][0][0]);
    for (int step = 0; step < NSTEPS; ++step) {
        const int kt = step * 32;
        if (w == 0) {
            if (step + 1 < NSTEPS) {
                tdm_load_tile64x32(wbase + kt + 32, &bt[(step + 1) & 1][0][0]);
                tdm_wait1();   // previous buffer's DMA done, next still in flight
            } else {
                tdm_wait0();
            }
        }
        __syncthreads();
        v16bf a = load_a_frag(arow + kt, half);
        if (kt + 32 < DM) __builtin_prefetch(arow + kt + 32, 0, 3);
        const __hip_bfloat16(*btc)[32] = bt[step & 1];
#pragma unroll
        for (int g = 0; g < 4; ++g)
            acc[g] = wmma_bf16(a, load_b_frag(&btc[16 * g + l16][0], half), acc[g]);
        __syncthreads();
    }

    const int h = blockIdx.y;  // 64-wide N tile == exactly one head
    if (z < 2) {
        __hip_bfloat16* dst = (z == 0) ? qout : kout;
#pragma unroll
        for (int g = 0; g < 4; ++g) {
            int d = 16 * g + l16;
            float inv = __expf(-(float)(d & ~1) * (9.210340372f / 64.0f)); // theta^(-2i/64)
#pragma unroll
            for (int j = 0; j < 8; ++j) {
                int m = mt + j + 8 * half;
                int s = m & (S_LEN - 1);
                int bb = m >> 11;
                float ang = (float)posn[s] * inv;
                float cs = __cosf(ang), sn = __sinf(ang);
                float self = acc[g][j];
                float partner = __shfl_xor(self, 1, 32);
                float val = ((d & 1) == 0) ? (self * cs - partner * sn)
                                           : (partner * sn + self * cs);
                dst[(((size_t)bb * NH + h) * S_LEN + s) * DH + d] = __float2bfloat16(val);
            }
        }
    } else {
#pragma unroll
        for (int g = 0; g < 4; ++g) {
            int d = 16 * g + l16;
#pragma unroll
            for (int j = 0; j < 8; ++j) {
                int m = mt + j + 8 * half;
                int s = m & (S_LEN - 1);
                int bb = m >> 11;
                vtout[(((size_t)bb * NH + h) * DH + d) * S_LEN + s] =
                    __float2bfloat16(acc[g][j]);
            }
        }
    }
}

// ---------------------------------------------------------------- flash attention
// grid: (128/4, B*NH); block 128 (4 waves). Each wave: one 16-query tile,
// 32-key steps, online softmax, P relayout via per-wave LDS tile.
__global__ __launch_bounds__(128) void attn_kernel(
    const __hip_bfloat16* __restrict__ q,
    const __hip_bfloat16* __restrict__ k,
    const __hip_bfloat16* __restrict__ vt,
    __hip_bfloat16* __restrict__ attn) {
    __shared__ __align__(16) __hip_bfloat16 pbuf[4][16][32];
    const int lane = threadIdx.x & 31;
    const int half = lane >> 4;
    const int l16  = lane & 15;
    const int w    = threadIdx.x >> 5;
    const int qt   = blockIdx.x * 4 + w;
    const int bh   = blockIdx.y;
    const size_t qk_base = (size_t)bh * S_LEN * DH;
    const size_t vt_base = (size_t)bh * DH * S_LEN;

    const __hip_bfloat16* qrow = q + qk_base + (size_t)(qt * 16 + l16) * DH;
    v16bf aq0 = load_a_frag(qrow, half);
    v16bf aq1 = load_a_frag(qrow + 32, half);

    v8f o[4];
#pragma unroll
    for (int g = 0; g < 4; ++g) o[g] = v8f_zero();
    float mrow[8], lrow[8];
#pragma unroll
    for (int j = 0; j < 8; ++j) { mrow[j] = -1e30f; lrow[j] = 0.0f; }

    const int qend = qt * 16 + 15;
    const int nsteps = (qend >> 5) + 1;
    for (int st = 0; st < nsteps; ++st) {
        const int kt = st * 32;
        v8f s0 = v8f_zero(), s1 = v8f_zero();
        const __hip_bfloat16* krow0 = k + qk_base + (size_t)(kt + l16) * DH;
        const __hip_bfloat16* krow1 = k + qk_base + (size_t)(kt + 16 + l16) * DH;
        s0 = wmma_bf16(aq0, load_b_frag(krow0, half), s0);
        s0 = wmma_bf16(aq1, load_b_frag(krow0 + 32, half), s0);
        s1 = wmma_bf16(aq0, load_b_frag(krow1, half), s1);
        s1 = wmma_bf16(aq1, load_b_frag(krow1 + 32, half), s1);

        const int key0 = kt + l16;
        const int key1 = kt + 16 + l16;
#pragma unroll
        for (int j = 0; j < 8; ++j) {
            int qpos = qt * 16 + j + 8 * half;
            float v0 = s0[j] * 0.125f;  // 1/sqrt(64)
            float v1 = s1[j] * 0.125f;
            if (key0 > qpos) v0 = -1e30f;
            if (key1 > qpos) v1 = -1e30f;
            float rm = fmaxf(v0, v1);
#pragma unroll
            for (int off = 1; off < 16; off <<= 1)
                rm = fmaxf(rm, __shfl_xor(rm, off, 32));
            float mnew = fmaxf(mrow[j], rm);
            float corr = __expf(mrow[j] - mnew);
            float p0 = __expf(v0 - mnew);
            float p1 = __expf(v1 - mnew);
            float rs = p0 + p1;
#pragma unroll
            for (int off = 1; off < 16; off <<= 1)
                rs += __shfl_xor(rs, off, 32);
            lrow[j] = lrow[j] * corr + rs;
            mrow[j] = mnew;
#pragma unroll
            for (int g = 0; g < 4; ++g) o[g][j] *= corr;
            pbuf[w][j + 8 * half][l16]      = __float2bfloat16(p0);
            pbuf[w][j + 8 * half][16 + l16] = __float2bfloat16(p1);
        }
        v16bf pa = load_a_frag(&pbuf[w][l16][0], half);
#pragma unroll
        for (int g = 0; g < 4; ++g) {
            const __hip_bfloat16* vrow = vt + vt_base + (size_t)(16 * g + l16) * S_LEN + kt;
            o[g] = wmma_bf16(pa, load_b_frag(vrow, half), o[g]);
        }
    }

    const int b = bh >> 4, h = bh & 15;
#pragma unroll
    for (int g = 0; g < 4; ++g) {
#pragma unroll
        for (int j = 0; j < 8; ++j) {
            int s = qt * 16 + j + 8 * half;
            float val = o[g][j] / lrow[j];
            attn[((size_t)(b * S_LEN + s)) * DM + h * DH + 16 * g + l16] =
                __float2bfloat16(val);
        }
    }
}

// ---------------------------------------------------------------- output proj
__global__ __launch_bounds__(256) void outproj_kernel(
    const __hip_bfloat16* __restrict__ a_in,
    const __hip_bfloat16* __restrict__ wo,
    float* __restrict__ out) {
    __shared__ __align__(16) __hip_bfloat16 bt[2][64][32];
    const int lane = threadIdx.x & 31;
    const int half = lane >> 4;
    const int l16  = lane & 15;
    const int w    = threadIdx.x >> 5;
    const int mt   = blockIdx.x * 128 + w * 16;
    const int nt   = blockIdx.y * 64;
    const __hip_bfloat16* wbase = wo + (size_t)nt * DM;

    v8f acc[4];
#pragma unroll
    for (int g = 0; g < 4; ++g) acc[g] = v8f_zero();

    const __hip_bfloat16* arow = a_in + (size_t)(mt + l16) * DM;

    const int NSTEPS = DM / 32;
    if (w == 0) tdm_load_tile64x32(wbase, &bt[0][0][0]);
    for (int step = 0; step < NSTEPS; ++step) {
        const int kt = step * 32;
        if (w == 0) {
            if (step + 1 < NSTEPS) {
                tdm_load_tile64x32(wbase + kt + 32, &bt[(step + 1) & 1][0][0]);
                tdm_wait1();
            } else {
                tdm_wait0();
            }
        }
        __syncthreads();
        v16bf a = load_a_frag(arow + kt, half);
        const __hip_bfloat16(*btc)[32] = bt[step & 1];
#pragma unroll
        for (int g = 0; g < 4; ++g)
            acc[g] = wmma_bf16(a, load_b_frag(&btc[16 * g + l16][0], half), acc[g]);
        __syncthreads();
    }
#pragma unroll
    for (int g = 0; g < 4; ++g) {
#pragma unroll
        for (int j = 0; j < 8; ++j) {
            out[(size_t)(mt + j + 8 * half) * DM + nt + 16 * g + l16] = acc[g][j];
        }
    }
}

// ---------------------------------------------------------------- host
extern "C" void kernel_launch(void* const* d_in, const int* in_sizes, int n_in,
                              void* d_out, int out_size, void* d_ws, size_t ws_size,
                              hipStream_t stream) {
    (void)in_sizes; (void)n_in; (void)out_size; (void)ws_size;
    const float* x_f  = (const float*)d_in[0];
    const int*   pos  = (const int*)d_in[1];
    const float* Wq_f = (const float*)d_in[2];
    const float* Wk_f = (const float*)d_in[3];
    const float* Wv_f = (const float*)d_in[4];
    const float* Wo_f = (const float*)d_in[5];
    float* out = (float*)d_out;

    const size_t nX = (size_t)BATCH * S_LEN * DM;  // 4M elems
    const size_t nW = (size_t)DM * DM;             // 1M elems
    char* p = (char*)d_ws;
    auto alloc = [&](size_t elems) {
        __hip_bfloat16* r = (__hip_bfloat16*)p;
        p += elems * sizeof(__hip_bfloat16);
        return r;
    };
    __hip_bfloat16* x_bf    = alloc(nX);
    __hip_bfloat16* wq_bf   = alloc(nW);
    __hip_bfloat16* wk_bf   = alloc(nW);
    __hip_bfloat16* wv_bf   = alloc(nW);
    __hip_bfloat16* wo_bf   = alloc(nW);
    __hip_bfloat16* q_bf    = alloc(nX);
    __hip_bfloat16* k_bf    = alloc(nX);
    __hip_bfloat16* vt_bf   = alloc(nX);
    __hip_bfloat16* attn_bf = alloc(nX);

    cvt_f32_bf16<<<1024, 256, 0, stream>>>(x_f, x_bf, (int)nX);
    cvt_f32_bf16<<<512, 256, 0, stream>>>(Wq_f, wq_bf, (int)nW);
    cvt_f32_bf16<<<512, 256, 0, stream>>>(Wk_f, wk_bf, (int)nW);
    cvt_f32_bf16<<<512, 256, 0, stream>>>(Wv_f, wv_bf, (int)nW);
    cvt_f32_bf16<<<512, 256, 0, stream>>>(Wo_f, wo_bf, (int)nW);

    qkv_rope_kernel<<<dim3(32, 16, 3), 256, 0, stream>>>(
        x_bf, wq_bf, wk_bf, wv_bf, pos, q_bf, k_bf, vt_bf);

    attn_kernel<<<dim3(32, 32), 128, 0, stream>>>(q_bf, k_bf, vt_bf, attn_bf);

    outproj_kernel<<<dim3(32, 16), 256, 0, stream>>>(attn_bf, wo_bf, out);
}
